// ResidualGAT_63007170232684
// MI455X (gfx1250) — compile-verified, matrix operand order checked by reading, not run on hardware
//
#include <hip/hip_runtime.h>

#define LRELU_SLOPE 0.2f

typedef __bf16 bf16;
typedef __attribute__((ext_vector_type(16))) __bf16 v16bf;
typedef __attribute__((ext_vector_type(8)))  __bf16 v8bf;
typedef __attribute__((ext_vector_type(8)))  float  v8f;

__device__ __forceinline__ unsigned short f2bf(float f) {
  unsigned u = __float_as_uint(f);
  u += 0x7FFFu + ((u >> 16) & 1u);         // round-to-nearest-even
  return (unsigned short)(u >> 16);
}
// monotone float<->uint key for atomic max over floats
__device__ __forceinline__ unsigned fkey(float f) {
  unsigned u = __float_as_uint(f);
  return (u >> 31) ? ~u : (u | 0x80000000u);
}
__device__ __forceinline__ float kdec(unsigned k) {
  unsigned u = (k >> 31) ? (k & 0x7FFFFFFFu) : ~k;
  return __uint_as_float(u);
}
__device__ __forceinline__ float leaky(float x) { return x > 0.f ? x : LRELU_SLOPE * x; }

// ---------------- bf16 cast ----------------
__global__ void cast_bf16_kernel(const float* __restrict__ in,
                                 unsigned short* __restrict__ out, int n) {
  int i = blockIdx.x * blockDim.x + threadIdx.x;
  if (i < n) out[i] = f2bf(in[i]);
}

// ------- pack [Wa | Wb] (Kin x half each, row-major) into WMMA B-operand layout -------
// bpack flat index = ((tile_n*kblk + kb)*32 + lane)*16 + elem
// lane<16 : N=lane,    elems 0..7 -> K kb*32+0..7,  elems 8..15 -> K kb*32+16..23
// lane>=16: N=lane-16, elems 0..7 -> K kb*32+8..15, elems 8..15 -> K kb*32+24..31
__global__ void pack_w_kernel(const float* __restrict__ Wa, const float* __restrict__ Wb,
                              int Kin, int half, unsigned short* __restrict__ bp) {
  int Kout = half * 2;
  int total = Kin * Kout;
  int idx = blockIdx.x * blockDim.x + threadIdx.x;
  if (idx >= total) return;
  int elem = idx & 15;
  int lane = (idx >> 4) & 31;
  int t    = idx >> 9;
  int kblk = Kin >> 5;
  int kb     = t % kblk;
  int tile_n = t / kblk;
  int ncol = tile_n * 16 + (lane & 15);
  int k = kb * 32 + (elem & 7) + ((elem >> 3) << 4) + ((lane >> 4) << 3);
  float v = (ncol < half) ? Wa[k * half + ncol] : Wb[k * half + (ncol - half)];
  bp[idx] = f2bf(v);
}

// ---------------- WMMA bf16 GEMM: C[MxKout] = A[MxKin] * Bpack ----------------
// one wave -> one 16x16 output tile; M % 16 == 0, Kin % 32 == 0, Kout % 16 == 0
__global__ void gemm_wmma_bf16_kernel(const unsigned short* __restrict__ A,
                                      const unsigned short* __restrict__ Bp,
                                      float* __restrict__ C,
                                      int M, int Kin, int Kout) {
  int kblk    = Kin >> 5;
  int tiles_n = Kout >> 4;
  int wave  = blockIdx.x * (blockDim.x >> 5) + (threadIdx.x >> 5);
  int total = (M >> 4) * tiles_n;
  if (wave >= total) return;                       // wave-uniform: EXEC stays all-ones
  int tile_m = wave / tiles_n;
  int tile_n = wave - tile_m * tiles_n;
  int lane = threadIdx.x & 31;
  int lh   = lane >> 4;                            // lane half selects K sub-block
  const bf16* arow = (const bf16*)A + (size_t)(tile_m * 16 + (lane & 15)) * Kin + (lh << 3);
  const bf16* bptr = (const bf16*)Bp + (size_t)tile_n * kblk * 512 + lane * 16;
  v8f acc = {};
  for (int kb = 0; kb < kblk; ++kb) {
    v8bf a0 = *(const v8bf*)(arow + kb * 32);      // K +0..7   (lane<16) / +8..15  (lane>=16)
    v8bf a1 = *(const v8bf*)(arow + kb * 32 + 16); // K +16..23 (lane<16) / +24..31 (lane>=16)
    union { v16bf v; v8bf h[2]; } u;
    u.h[0] = a0; u.h[1] = a1;
    v16bf bv = *(const v16bf*)(bptr + kb * 512);
    acc = __builtin_amdgcn_wmma_f32_16x16x32_bf16(false, u.v, false, bv,
                                                  (short)0, acc, false, false);
  }
  // D layout: VGPR r -> row (tile_m*16 + r + lh*8), col = tile_n*16 + (lane&15)
  size_t base = (size_t)(tile_m * 16 + (lh << 3)) * Kout + tile_n * 16 + (lane & 15);
  #pragma unroll
  for (int r = 0; r < 8; ++r) C[base + (size_t)r * Kout] = acc[r];
}

// ---------------- per-node attention dots: es/ed [N,16] ----------------
__global__ void attn_node_kernel(const float* __restrict__ h,
                                 const float* __restrict__ asA, const float* __restrict__ asB,
                                 const float* __restrict__ adA, const float* __restrict__ adB,
                                 float* __restrict__ es, float* __restrict__ ed,
                                 int Nn, int C) {
  int idx = blockIdx.x * blockDim.x + threadIdx.x;  // n*16 + head
  if (idx >= Nn * 16) return;
  int head = idx & 15;
  const float* hv = h + (size_t)(idx >> 4) * (16 * C) + head * C;
  const float* as = (head < 8) ? asA + head * C : asB + (head - 8) * C;
  const float* ad = (head < 8) ? adA + head * C : adB + (head - 8) * C;
  float s = 0.f, d = 0.f;
  for (int c = 0; c < C; ++c) { float v = hv[c]; s += v * as[c]; d += v * ad[c]; }
  es[idx] = s; ed[idx] = d;
}

// ---------------- edge pass 1: segment max (as uint keys) ----------------
__global__ void edge_max_kernel(const int* __restrict__ ei, int E, int Etot,
                                const float* __restrict__ es, const float* __restrict__ ed,
                                unsigned* __restrict__ mk) {
  int idx = blockIdx.x * blockDim.x + threadIdx.x;  // e*16 + head
  int e = idx >> 4;
  if (e >= Etot) return;
  int hh = idx & 15;
  int src, dst;
  if (e < E) { src = ei[e]; dst = ei[E + e]; } else { src = dst = e - E; }
  float att = leaky(es[src * 16 + hh] + ed[dst * 16 + hh]);
  atomicMax(&mk[dst * 16 + hh], fkey(att));
}

// ---------------- edge pass 2: exp-sum ----------------
__global__ void edge_expsum_kernel(const int* __restrict__ ei, int E, int Etot,
                                   const float* __restrict__ es, const float* __restrict__ ed,
                                   const unsigned* __restrict__ mk, float* __restrict__ dn) {
  int idx = blockIdx.x * blockDim.x + threadIdx.x;
  int e = idx >> 4;
  if (e >= Etot) return;
  int hh = idx & 15;
  int src, dst;
  if (e < E) { src = ei[e]; dst = ei[E + e]; } else { src = dst = e - E; }
  float att = leaky(es[src * 16 + hh] + ed[dst * 16 + hh]);
  atomicAdd(&dn[dst * 16 + hh], __expf(att - kdec(mk[dst * 16 + hh])));
}

// ---------------- edge pass 3: weighted scatter-add (wave per edge) ----------------
template <int C>
__global__ void edge_aggr_kernel(const int* __restrict__ ei, int E, int Etot,
                                 const float* __restrict__ es, const float* __restrict__ ed,
                                 const unsigned* __restrict__ mk, const float* __restrict__ dn,
                                 const float* __restrict__ h, float* __restrict__ out) {
  int wid = blockIdx.x * (blockDim.x >> 5) + (threadIdx.x >> 5);
  if (wid >= Etot) return;
  int lane = threadIdx.x & 31;
  int src, dst;
  if (wid < E) { src = ei[wid]; dst = ei[E + wid]; } else { src = dst = wid - E; }
  const int per = C >> 1;                 // channels per lane (lane pair covers one head)
  int head = lane >> 1;
  int coff = (lane & 1) * per;
  float att   = leaky(es[src * 16 + head] + ed[dst * 16 + head]);
  float alpha = __expf(att - kdec(mk[dst * 16 + head])) / dn[dst * 16 + head];
  const int HC = 16 * C;
  const float* hs = h   + (size_t)src * HC + head * C + coff;
  float*       od = out + (size_t)dst * HC + head * C + coff;
  #pragma unroll
  for (int j = 0; j < per; ++j) atomicAdd(&od[j], hs[j] * alpha);
}

// ---------------- layer-1 epilogue: elu(conv)+elu(res) -> x1 (f32 + bf16) ----------------
__global__ void finalize1_kernel(const float* __restrict__ oc,
                                 const float* __restrict__ b1, const float* __restrict__ b1r,
                                 float* __restrict__ x1, unsigned short* __restrict__ x1bf,
                                 int Nn) {
  int idx = blockIdx.x * blockDim.x + threadIdx.x;   // n*64 + c
  if (idx >= Nn * 64) return;
  int n = idx >> 6, c = idx & 63;
  float va = oc[(size_t)n * 128 + c]      + b1[c];
  float vb = oc[(size_t)n * 128 + 64 + c] + b1r[c];
  va = va > 0.f ? va : __expf(va) - 1.f;
  vb = vb > 0.f ? vb : __expf(vb) - 1.f;
  float v = va + vb;
  x1[idx]   = v;
  x1bf[idx] = f2bf(v);
}

// ---------------- layer-2 epilogue: head-mean + bias + log_softmax ----------------
__global__ void finalize2_kernel(const float* __restrict__ oc,
                                 const float* __restrict__ b2, const float* __restrict__ b2r,
                                 float* __restrict__ out, int Nn) {
  int n = blockIdx.x * blockDim.x + threadIdx.x;
  if (n >= Nn) return;
  const float* row = oc + (size_t)n * 256;
  float logits[16];
  #pragma unroll
  for (int c = 0; c < 16; ++c) {
    float sa = 0.f, sb = 0.f;
    #pragma unroll
    for (int hh = 0; hh < 8; ++hh) { sa += row[hh * 16 + c]; sb += row[128 + hh * 16 + c]; }
    logits[c] = sa * 0.125f + b2[c] + sb * 0.125f + b2r[c];
  }
  float mx = logits[0];
  #pragma unroll
  for (int c = 1; c < 16; ++c) mx = fmaxf(mx, logits[c]);
  float s = 0.f;
  #pragma unroll
  for (int c = 0; c < 16; ++c) s += __expf(logits[c] - mx);
  float lse = __logf(s) + mx;
  #pragma unroll
  for (int c = 0; c < 16; ++c) out[(size_t)n * 16 + c] = logits[c] - lse;
}

extern "C" void kernel_launch(void* const* d_in, const int* in_sizes, int n_in,
                              void* d_out, int out_size, void* d_ws, size_t ws_size,
                              hipStream_t stream) {
  (void)n_in; (void)out_size; (void)ws_size;
  const float* x       = (const float*)d_in[0];
  const int*   ei      = (const int*)d_in[1];
  const float* W1      = (const float*)d_in[2];
  const float* a_src1  = (const float*)d_in[3];
  const float* a_dst1  = (const float*)d_in[4];
  const float* b1      = (const float*)d_in[5];
  const float* W1r     = (const float*)d_in[6];
  const float* a_src1r = (const float*)d_in[7];
  const float* a_dst1r = (const float*)d_in[8];
  const float* b1r     = (const float*)d_in[9];
  const float* W2      = (const float*)d_in[10];
  const float* a_src2  = (const float*)d_in[11];
  const float* a_dst2  = (const float*)d_in[12];
  const float* b2      = (const float*)d_in[13];
  const float* W2r     = (const float*)d_in[14];
  const float* a_src2r = (const float*)d_in[15];
  const float* a_dst2r = (const float*)d_in[16];
  const float* b2r     = (const float*)d_in[17];

  const int FIN = 128;
  const int N    = in_sizes[0] / FIN;   // 100000 (divisible by 16)
  const int E    = in_sizes[1] / 2;     // 1600000
  const int Etot = E + N;               // + self loops

  // ---- workspace carve-out (~295 MB) ----
  char* p = (char*)d_ws;
  auto take = [&](size_t bytes) -> char* {
    char* r = p; p += (bytes + 255) & ~(size_t)255; return r;
  };
  unsigned short* xbf  = (unsigned short*)take((size_t)N * 128 * 2);
  unsigned short* wp1  = (unsigned short*)take(128 * 128 * 2);
  unsigned short* wp2  = (unsigned short*)take(64 * 256 * 2);
  float*    R1   = (float*)take((size_t)N * 256 * 4);   // h1 [Nx128] -> out2 [Nx256]
  float*    R2   = (float*)take((size_t)N * 256 * 4);   // out1 [Nx128] -> h2 [Nx256]
  float*    es   = (float*)take((size_t)N * 16 * 4);
  float*    ed   = (float*)take((size_t)N * 16 * 4);
  unsigned* mk   = (unsigned*)take((size_t)N * 16 * 4);
  float*    dn   = (float*)take((size_t)N * 16 * 4);
  float*    x1   = (float*)take((size_t)N * 64 * 4);
  unsigned short* x1bf = (unsigned short*)take((size_t)N * 64 * 2);

  const int T = 256;  // 8 waves per block on wave32
  auto blk  = [T](long long work) { return (unsigned)((work + T - 1) / T); };

  // precision/prep
  cast_bf16_kernel<<<blk((long long)N * 128), T, 0, stream>>>(x, xbf, N * 128);
  pack_w_kernel<<<blk(128 * 128), T, 0, stream>>>(W1, W1r, 128, 64, wp1);
  pack_w_kernel<<<blk(64 * 256),  T, 0, stream>>>(W2, W2r, 64, 128, wp2);

  // ===== layer 1 (conv1 + res fused as 16 heads x 8 ch) =====
  gemm_wmma_bf16_kernel<<<(unsigned)(((N / 16) * 8 + 7) / 8), T, 0, stream>>>(
      xbf, wp1, R1, N, 128, 128);
  attn_node_kernel<<<blk((long long)N * 16), T, 0, stream>>>(
      R1, a_src1, a_src1r, a_dst1, a_dst1r, es, ed, N, 8);
  hipMemsetAsync(mk, 0, (size_t)N * 16 * 4, stream);
  hipMemsetAsync(dn, 0, (size_t)N * 16 * 4, stream);
  hipMemsetAsync(R2, 0, (size_t)N * 128 * 4, stream);
  edge_max_kernel<<<blk((long long)Etot * 16), T, 0, stream>>>(ei, E, Etot, es, ed, mk);
  edge_expsum_kernel<<<blk((long long)Etot * 16), T, 0, stream>>>(ei, E, Etot, es, ed, mk, dn);
  edge_aggr_kernel<8><<<(unsigned)((Etot + 7) / 8), T, 0, stream>>>(
      ei, E, Etot, es, ed, mk, dn, R1, R2);
  finalize1_kernel<<<blk((long long)N * 64), T, 0, stream>>>(R2, b1, b1r, x1, x1bf, N);

  // ===== layer 2 (conv2 + res fused as 16 heads x 16 ch) =====
  gemm_wmma_bf16_kernel<<<(unsigned)(((N / 16) * 16 + 7) / 8), T, 0, stream>>>(
      x1bf, wp2, R2, N, 64, 256);
  attn_node_kernel<<<blk((long long)N * 16), T, 0, stream>>>(
      R2, a_src2, a_src2r, a_dst2, a_dst2r, es, ed, N, 16);
  hipMemsetAsync(mk, 0, (size_t)N * 16 * 4, stream);
  hipMemsetAsync(dn, 0, (size_t)N * 16 * 4, stream);
  hipMemsetAsync(R1, 0, (size_t)N * 256 * 4, stream);
  edge_max_kernel<<<blk((long long)Etot * 16), T, 0, stream>>>(ei, E, Etot, es, ed, mk);
  edge_expsum_kernel<<<blk((long long)Etot * 16), T, 0, stream>>>(ei, E, Etot, es, ed, mk, dn);
  edge_aggr_kernel<16><<<(unsigned)((Etot + 7) / 8), T, 0, stream>>>(
      ei, E, Etot, es, ed, mk, dn, R2, R1);
  finalize2_kernel<<<blk(N), T, 0, stream>>>(R1, b2, b2r, (float*)d_out, N);
}
